// CaptioningRNN_91036126806920
// MI455X (gfx1250) — compile-verified, matrix operand order checked by reading, not run on hardware
//
#include <hip/hip_runtime.h>
#include <math.h>

typedef __attribute__((ext_vector_type(16))) __bf16 v16bf;
typedef __attribute__((ext_vector_type(8)))  float  v8f;

#define N_  128
#define T_  64
#define D_  512
#define H_  1024
#define L_  196              // 14*14
#define K_  2560             // D + H + H
#define G_  4096             // 4*H
#define KT_ 80               // K_/32
#define HL_ (H_*L_)

// ---------- bf16 helpers (manual RNE, no __hip_bfloat16 arithmetic) ----------
__device__ __forceinline__ unsigned short f32_bf16(float f) {
  union { float f; unsigned u; } x; x.f = f;
  unsigned u = x.u;
  u += 0x7FFFu + ((u >> 16) & 1u);
  return (unsigned short)(u >> 16);
}
__device__ __forceinline__ float bf16_f32(unsigned short s) {
  union { unsigned u; float f; } x; x.u = ((unsigned)s) << 16;
  return x.f;
}
__device__ __forceinline__ unsigned pack2(float a, float b) {
  return (unsigned)f32_bf16(a) | ((unsigned)f32_bf16(b) << 16);
}

union FragU { uint4 q[2]; v16bf v; };

// ---------- prep: A -> bf16 copy, h0 = c0 = mean over L ----------
__global__ void prep_A(const float* __restrict__ A, unsigned short* __restrict__ Abf,
                       float* __restrict__ h, float* __restrict__ c) {
  int idx = blockIdx.x * blockDim.x + threadIdx.x;      // over N*H
  const float* src = A + (size_t)idx * L_;
  unsigned short* dst = Abf + (size_t)idx * L_;
  float s = 0.f;
  #pragma unroll 4
  for (int l = 0; l < L_; ++l) { float v = src[l]; s += v; dst[l] = f32_bf16(v); }
  float m = s * (1.0f / (float)L_);
  h[idx] = m; c[idx] = m;
}

// ---------- prep: pack [Wx;Wh;Wattn] into WMMA B-fragment layout ----------
// Word layout: Wp[((nt*KT + kt)*32 + lane)*8 + v], each word = bf16 pair (k_even lo, k_odd hi)
// 16-bit 16x32 fragment K-mapping: lanes 0-15 -> K 0..7 (v0..3), 16..23 (v4..7);
//                                  lanes 16-31 -> K 8..15, 24..31.
__global__ void pack_W(const float* __restrict__ Wx, const float* __restrict__ Wh,
                       const float* __restrict__ Wattn, unsigned* __restrict__ Wp) {
  int tid = blockIdx.x * blockDim.x + threadIdx.x;      // K_*G_/2 = 5,242,880 words
  int v    = tid & 7;
  int lane = (tid >> 3) & 31;
  int kt   = (tid >> 8) % KT_;
  int nt   = (tid >> 8) / KT_;
  int n    = nt * 16 + (lane & 15);
  int half = lane >> 4;
  int kb   = (v < 4) ? (v * 2 + half * 8) : (16 + (v - 4) * 2 + half * 8);
  int k0   = kt * 32 + kb;                              // even; k0+1 same source region
  float f0, f1;
  if (k0 < D_)            { const float* p = Wx    + (size_t)k0 * G_ + n;             f0 = p[0]; f1 = p[G_]; }
  else if (k0 < D_ + H_)  { const float* p = Wh    + (size_t)(k0 - D_) * G_ + n;      f0 = p[0]; f1 = p[G_]; }
  else                    { const float* p = Wattn + (size_t)(k0 - D_ - H_) * G_ + n; f0 = p[0]; f1 = p[G_]; }
  Wp[tid] = pack2(f0, f1);
}

// ---------- per-step: attention + pack Z = [x_t | h | attn] into A-fragment layout ----------
__global__ void __launch_bounds__(256) attention_step(
    const float* __restrict__ x, const unsigned short* __restrict__ Abf,
    const float* __restrict__ h, unsigned* __restrict__ Zp, int t) {
  __shared__ float h_sh[H_];
  __shared__ float attn_sh[H_];
  __shared__ float w_sh[256];
  __shared__ float red[256];
  int n  = blockIdx.x;
  int ti = threadIdx.x;

  for (int i = ti; i < H_; i += 256) h_sh[i] = h[n * H_ + i];
  __syncthreads();

  // scores[l] = (1/sqrt(H)) * sum_h h[h]*A[n,h,l]  — lane-adjacent l => coalesced
  const unsigned short* An = Abf + (size_t)n * HL_;
  float sv = -3.0e38f;
  if (ti < L_) {
    float s = 0.f;
    const unsigned short* p = An + ti;
    #pragma unroll 4
    for (int hh = 0; hh < H_; ++hh) s += h_sh[hh] * bf16_f32(p[hh * L_]);
    sv = s * 0.03125f;                                  // 1/sqrt(1024)
    w_sh[ti] = sv;
  }
  __syncthreads();

  // softmax over L
  red[ti] = sv; __syncthreads();
  for (int s = 128; s > 0; s >>= 1) { if (ti < s) red[ti] = fmaxf(red[ti], red[ti + s]); __syncthreads(); }
  float mx = red[0]; __syncthreads();
  float ev = (ti < L_) ? __expf(sv - mx) : 0.f;
  red[ti] = ev; __syncthreads();
  for (int s = 128; s > 0; s >>= 1) { if (ti < s) red[ti] += red[ti + s]; __syncthreads(); }
  float inv = 1.0f / red[0]; __syncthreads();
  if (ti < L_) w_sh[ti] = ev * inv;
  __syncthreads();

  // attn[h] = sum_l A[n,h,l]*w[l] — wave per h-chunk, lanes split l, wave reduce
  int wv = ti >> 5, lane = ti & 31;
  for (int hu = wv * 128; hu < wv * 128 + 128; ++hu) {
    const unsigned short* q = An + hu * L_;
    float a = 0.f;
    for (int l = lane; l < L_; l += 32) a += bf16_f32(q[l]) * w_sh[l];
    #pragma unroll
    for (int off = 16; off > 0; off >>= 1) a += __shfl_down(a, off, 32);
    if (lane == 0) attn_sh[hu] = a;
  }
  __syncthreads();

  // pack this sample's row of Z into WMMA A-fragment layout
  int mt = n >> 4, M = n & 15;
  for (int p = ti; p < K_ / 2; p += 256) {
    int k0 = p * 2;
    float v0, v1;
    if (k0 < D_)           { const float* xp = x + ((size_t)n * T_ + t) * D_ + k0; v0 = xp[0]; v1 = xp[1]; }
    else if (k0 < D_ + H_) { v0 = h_sh[k0 - D_];          v1 = h_sh[k0 - D_ + 1]; }
    else                   { v0 = attn_sh[k0 - D_ - H_];  v1 = attn_sh[k0 - D_ - H_ + 1]; }
    int kt   = k0 >> 5, kp = k0 & 31;
    int half = (kp >> 3) & 1;
    int vg   = ((kp & 7) >> 1) + ((kp >> 4) << 2);
    int ln   = M + (half << 4);
    Zp[((mt * KT_ + kt) * 32 + ln) * 8 + vg] = pack2(v0, v1);
  }
}

// ---------- per-step: gates GEMM  a = Z(128x2560) @ W(2560x4096), bf16 WMMA / f32 acc ----------
// One wave -> 16x64 output strip (1 A-frag reused across 4 B-frags).
// Software-pipelined: fragments for kt+1 are fetched while the 4 WMMAs for kt issue,
// so the per-iteration s_wait covers loads issued a full iteration earlier.
__global__ void __launch_bounds__(256) gemm_gates(
    const unsigned* __restrict__ Zp, const unsigned* __restrict__ Wp,
    float* __restrict__ apre) {
  int wid  = blockIdx.x * 8 + (threadIdx.x >> 5);       // 512 waves
  int lane = threadIdx.x & 31;
  int mt   = wid >> 6;                                  // 0..7  (M tile)
  int ntg  = wid & 63;                                  // 0..63 (group of 4 N tiles)
  int nt0  = ntg * 4;

  // per-kt stride = 32 lanes * 8 dwords = 64 uint4
  const uint4* zb = (const uint4*)Zp + ((size_t)mt        * KT_ * 32 + lane) * 2;
  const uint4* w0 = (const uint4*)Wp + ((size_t)(nt0 + 0) * KT_ * 32 + lane) * 2;
  const uint4* w1 = (const uint4*)Wp + ((size_t)(nt0 + 1) * KT_ * 32 + lane) * 2;
  const uint4* w2 = (const uint4*)Wp + ((size_t)(nt0 + 2) * KT_ * 32 + lane) * 2;
  const uint4* w3 = (const uint4*)Wp + ((size_t)(nt0 + 3) * KT_ * 32 + lane) * 2;

  v8f acc0 = {0.f,0.f,0.f,0.f,0.f,0.f,0.f,0.f};
  v8f acc1 = acc0, acc2 = acc0, acc3 = acc0;

  FragU a, b0, b1, b2, b3;                              // current (kt)
  a.q[0]  = zb[0]; a.q[1]  = zb[1];
  b0.q[0] = w0[0]; b0.q[1] = w0[1];
  b1.q[0] = w1[0]; b1.q[1] = w1[1];
  b2.q[0] = w2[0]; b2.q[1] = w2[1];
  b3.q[0] = w3[0]; b3.q[1] = w3[1];

  for (int kt = 0; kt < KT_ - 1; ++kt) {
    size_t off = (size_t)(kt + 1) * 64;                 // next kt, uint4 units
    FragU an, bn0, bn1, bn2, bn3;                       // prefetch (kt+1)
    an.q[0]  = zb[off];     an.q[1]  = zb[off + 1];
    bn0.q[0] = w0[off];     bn0.q[1] = w0[off + 1];
    bn1.q[0] = w1[off];     bn1.q[1] = w1[off + 1];
    bn2.q[0] = w2[off];     bn2.q[1] = w2[off + 1];
    bn3.q[0] = w3[off];     bn3.q[1] = w3[off + 1];

    acc0 = __builtin_amdgcn_wmma_f32_16x16x32_bf16(false, a.v, false, b0.v, (short)0, acc0, false, false);
    acc1 = __builtin_amdgcn_wmma_f32_16x16x32_bf16(false, a.v, false, b1.v, (short)0, acc1, false, false);
    acc2 = __builtin_amdgcn_wmma_f32_16x16x32_bf16(false, a.v, false, b2.v, (short)0, acc2, false, false);
    acc3 = __builtin_amdgcn_wmma_f32_16x16x32_bf16(false, a.v, false, b3.v, (short)0, acc3, false, false);

    a = an; b0 = bn0; b1 = bn1; b2 = bn2; b3 = bn3;
  }
  // epilogue: kt = KT_-1
  acc0 = __builtin_amdgcn_wmma_f32_16x16x32_bf16(false, a.v, false, b0.v, (short)0, acc0, false, false);
  acc1 = __builtin_amdgcn_wmma_f32_16x16x32_bf16(false, a.v, false, b1.v, (short)0, acc1, false, false);
  acc2 = __builtin_amdgcn_wmma_f32_16x16x32_bf16(false, a.v, false, b2.v, (short)0, acc2, false, false);
  acc3 = __builtin_amdgcn_wmma_f32_16x16x32_bf16(false, a.v, false, b3.v, (short)0, acc3, false, false);

  // C/D layout: lanes 0-15: N=lane, M=r; lanes 16-31: N=lane-16, M=r+8
  int col = lane & 15;
  int mrow = mt * 16 + (lane >> 4) * 8;
  #pragma unroll
  for (int r = 0; r < 8; ++r) {
    size_t row = (size_t)(mrow + r) * G_;
    apre[row + (nt0 + 0) * 16 + col] = acc0[r];
    apre[row + (nt0 + 1) * 16 + col] = acc1[r];
    apre[row + (nt0 + 2) * 16 + col] = acc2[r];
    apre[row + (nt0 + 3) * 16 + col] = acc3[r];
  }
}

// ---------- per-step: gate nonlinearities + state update ----------
__global__ void lstm_update(const float* __restrict__ apre, const float* __restrict__ b,
                            float* __restrict__ h, float* __restrict__ c,
                            float* __restrict__ out, int t) {
  int idx = blockIdx.x * blockDim.x + threadIdx.x;      // N*H
  int n = idx >> 10, u = idx & 1023;
  const float* an = apre + (size_t)n * G_;
  float ai = an[u]          + b[u];
  float af = an[H_ + u]     + b[H_ + u];
  float ao = an[2 * H_ + u] + b[2 * H_ + u];
  float ag = an[3 * H_ + u] + b[3 * H_ + u];
  float ig = 1.f / (1.f + __expf(-ai));
  float fg = 1.f / (1.f + __expf(-af));
  float og = 1.f / (1.f + __expf(-ao));
  float gg = tanhf(ag);
  float cn = fg * c[idx] + ig * gg;
  float hn = og * tanhf(cn);
  c[idx] = cn; h[idx] = hn;
  out[((size_t)n * T_ + t) * H_ + u] = hn;
}

extern "C" void kernel_launch(void* const* d_in, const int* in_sizes, int n_in,
                              void* d_out, int out_size, void* d_ws, size_t ws_size,
                              hipStream_t stream) {
  const float* x     = (const float*)d_in[0];
  const float* A     = (const float*)d_in[1];
  const float* Wx    = (const float*)d_in[2];
  const float* Wh    = (const float*)d_in[3];
  const float* Wattn = (const float*)d_in[4];
  const float* b     = (const float*)d_in[5];
  float* out = (float*)d_out;
  (void)in_sizes; (void)n_in; (void)out_size; (void)ws_size;

  char* w = (char*)d_ws;
  auto take = [&](size_t bytes) { char* p = w; w += (bytes + 255) & ~(size_t)255; return p; };
  unsigned short* Abf = (unsigned short*)take((size_t)N_ * H_ * L_ * 2);   // 51.4 MB
  unsigned*       Wp  = (unsigned*)      take((size_t)K_ * G_ / 2 * 4);    // 21.0 MB
  unsigned*       Zp  = (unsigned*)      take((size_t)N_ * K_ / 2 * 4);    // 0.66 MB
  float*          hb  = (float*)         take((size_t)N_ * H_ * 4);
  float*          cb  = (float*)         take((size_t)N_ * H_ * 4);
  float*          ap  = (float*)         take((size_t)N_ * G_ * 4);        // 2.1 MB

  prep_A<<<(N_ * H_) / 256, 256, 0, stream>>>(A, Abf, hb, cb);
  pack_W<<<(K_ * G_ / 2) / 256, 256, 0, stream>>>(Wx, Wh, Wattn, Wp);
  for (int t = 0; t < T_; ++t) {
    attention_step<<<N_, 256, 0, stream>>>(x, Abf, hb, Zp, t);
    gemm_gates<<<64, 256, 0, stream>>>(Zp, Wp, ap);
    lstm_update<<<(N_ * H_) / 256, 256, 0, stream>>>(ap, b, hb, cb, out, t);
  }
}
